// HierarchicalMambaBGNN_21526376087908
// MI455X (gfx1250) — compile-verified, matrix operand order checked by reading, not run on hardware
//
#include <hip/hip_runtime.h>
#include <hip/hip_bf16.h>

// ---------------- model constants ----------------
constexpr int NB   = 16;   // batch
constexpr int NT   = 512;  // seq len
constexpr int ND   = 256;  // d_model / nodes
constexpr int NHID = 256;
constexpr int NGS  = 4;
constexpr int NSS  = 6;
constexpr int NAS  = 4;
constexpr int NH   = 4;    // heads
constexpr int NMC  = 4;    // MC samples
constexpr int NDE  = 10;
constexpr int NDH  = 128;
constexpr int NKH  = 3;    // hops
constexpr int NSTEPS = 3;
constexpr int NDECH  = 128;

typedef __attribute__((ext_vector_type(16))) __bf16 v16bf;
typedef __attribute__((ext_vector_type(8)))  __bf16 v8bf;
typedef __attribute__((ext_vector_type(8)))  float  v8f;

// ---------------- helpers ----------------
__device__ __forceinline__ unsigned pcg(unsigned x) {
  x = x * 747796405u + 2891336453u;
  unsigned w = ((x >> ((x >> 28u) + 4u)) ^ x) * 277803737u;
  return (w >> 22u) ^ w;
}
__device__ __forceinline__ float u01(unsigned h) {
  return ((h >> 8) + 0.5f) * (1.0f / 16777216.0f);
}
__device__ __forceinline__ float gumbel_h(unsigned seed) {
  float u = u01(pcg(seed));
  u = fminf(fmaxf(u, 1e-6f), 1.0f - 1e-6f);
  return -logf(-logf(u));
}
__device__ __forceinline__ float gauss_h(unsigned seed) {
  float u1 = u01(pcg(seed));
  float u2 = u01(pcg(seed ^ 0x9e3779b9u));
  u1 = fmaxf(u1, 1e-12f);
  return sqrtf(-2.0f * logf(u1)) * cosf(6.28318530718f * u2);
}
__device__ __forceinline__ float gelu_f(float x) {
  return 0.5f * x * (1.0f + erff(x * 0.70710678118f));
}
__device__ __forceinline__ float sigm_f(float x) { return 1.0f / (1.0f + expf(-x)); }

template <int BS>
__device__ float block_sum(float v, float* red, int tid) {
  red[tid] = v; __syncthreads();
  for (int s = BS / 2; s > 0; s >>= 1) {
    if (tid < s) red[tid] += red[tid + s];
    __syncthreads();
  }
  float r = red[0]; __syncthreads();
  return r;
}
template <int BS>
__device__ float block_max(float v, float* red, int tid) {
  red[tid] = v; __syncthreads();
  for (int s = BS / 2; s > 0; s >>= 1) {
    if (tid < s) red[tid] = fmaxf(red[tid], red[tid + s]);
    __syncthreads();
  }
  float r = red[0]; __syncthreads();
  return r;
}

// ---------------- WMMA GEMM: C[m,n] = alpha * sum_k A[m,k]*Bt[n,k] (+bias[n]) ----------------
// Both operands are "rows x K" row-major, staged to LDS with identical padded
// [row][k] layout; every fragment is two contiguous ds_load_b128 per the CDNA5
// 16-bit A-matrix layout (k in [laneHi,laneHi+8) and [16+laneHi, 16+laneHi+8)).
// Block tile 64x128, 8 waves in 2x4, each wave 32x32 => 4 WMMA per k-step.
// Output modes:
//   default      : f32 row-major C (optionally += with GF_ACC)
//   GF_DUAL      : bf16 row-major C  AND transposed bf16 copy C2[n][m]
//                  (C2 written as packed 16B v8bf stores: 8 consecutive m)
//   GF_SPLITB    : with GF_DUAL: C2 index decodes m=b*512+t -> [b][n][t]
#define GF_ACC    1
#define GF_DUAL   2
#define GF_SPLITB 4

__device__ __forceinline__ v16bf ld_frag(const __bf16* rowp, int laneHi) {
  v16bf v;
  *(v8bf*)&v       = *(const v8bf*)(rowp + laneHi);
  *((v8bf*)&v + 1) = *(const v8bf*)(rowp + 16 + laneHi);
  return v;
}

__global__ __launch_bounds__(256)
void k_wmma_gemm(const __bf16* __restrict__ Ag, int lda, long sAz,
                 const __bf16* __restrict__ Btg, int ldb, long sBz,
                 void* __restrict__ Cg, int ldc, long sCz,
                 __bf16* __restrict__ C2g, int ld2, long sC2z,
                 const float* __restrict__ bias, float alpha,
                 int M, int N, int Kdim, int flags) {
  const int z = blockIdx.z;
  const __bf16* A  = Ag  + (long)z * sAz;
  const __bf16* Bt = Btg + (long)z * sBz;
  const int rowBase = blockIdx.y * 64;
  const int colBase = blockIdx.x * 128;
  __shared__ __align__(16) __bf16 As[64 * 40];   // [m][k], k padded 32->40
  __shared__ __align__(16) __bf16 Bs[128 * 40];  // [n][k], k padded 32->40
  const int tid  = threadIdx.x;
  const int lane = tid & 31;
  const int wave = tid >> 5;
  const int wr = wave >> 2, wc = wave & 3;       // 2x4 wave grid -> 64x128
  const int laneHi = (lane >> 4) << 3;           // 0 or 8 (K-half select)
  const int lm = lane & 15;
  v8f acc00 = {}, acc01 = {}, acc10 = {}, acc11 = {};
  const int arow = tid >> 2, acol = (tid & 3) << 3;  // A: 64 rows x 4 chunks
  for (int k0 = 0; k0 < Kdim; k0 += 32) {
    *(uint4*)(&As[arow * 40 + acol]) =
        *(const uint4*)(A + (long)(rowBase + arow) * lda + k0 + acol);
#pragma unroll
    for (int c = 0; c < 2; c++) {
      int chunk = tid + c * 256;                 // 512 chunks: 128 rows x 4
      int row = chunk >> 2, kg = (chunk & 3) << 3;
      *(uint4*)(&Bs[row * 40 + kg]) =
          *(const uint4*)(Bt + (long)(colBase + row) * ldb + k0 + kg);
    }
    if (k0 + 32 < Kdim) {
      __builtin_prefetch((const void*)(A + (long)(rowBase + arow) * lda + k0 + 32 + acol), 0, 0);
      __builtin_prefetch((const void*)(Bt + (long)(colBase + (tid >> 2)) * ldb + k0 + 32), 0, 0);
    }
    __syncthreads();
    v16bf av0 = ld_frag(&As[(wr * 32 + lm) * 40], laneHi);
    v16bf av1 = ld_frag(&As[(wr * 32 + lm + 16) * 40], laneHi);
    v16bf bv0 = ld_frag(&Bs[(wc * 32 + lm) * 40], laneHi);
    v16bf bv1 = ld_frag(&Bs[(wc * 32 + lm + 16) * 40], laneHi);
    acc00 = __builtin_amdgcn_wmma_f32_16x16x32_bf16(false, av0, false, bv0, (short)0, acc00, false, false);
    acc01 = __builtin_amdgcn_wmma_f32_16x16x32_bf16(false, av0, false, bv1, (short)0, acc01, false, false);
    acc10 = __builtin_amdgcn_wmma_f32_16x16x32_bf16(false, av1, false, bv0, (short)0, acc10, false, false);
    acc11 = __builtin_amdgcn_wmma_f32_16x16x32_bf16(false, av1, false, bv1, (short)0, acc11, false, false);
    __syncthreads();
  }
  const int mBase = rowBase + wr * 32 + ((lane >= 16) ? 8 : 0);
  const int n0 = colBase + wc * 32 + lm;
  const int n1 = n0 + 16;
  const float b0 = bias ? bias[n0] : 0.f;
  const float b1 = bias ? bias[n1] : 0.f;
  if (flags & GF_DUAL) {
    __bf16* C  = (__bf16*)Cg + (long)z * sCz;
    __bf16* C2 = C2g + (long)z * sC2z;
    v8bf p00, p01, p10, p11;
#pragma unroll
    for (int r = 0; r < 8; r++) {
      float v00 = alpha * acc00[r] + b0;
      float v01 = alpha * acc01[r] + b1;
      float v10 = alpha * acc10[r] + b0;
      float v11 = alpha * acc11[r] + b1;
      C[(long)(mBase + r) * ldc + n0]      = (__bf16)v00;
      C[(long)(mBase + r) * ldc + n1]      = (__bf16)v01;
      C[(long)(mBase + 16 + r) * ldc + n0] = (__bf16)v10;
      C[(long)(mBase + 16 + r) * ldc + n1] = (__bf16)v11;
      p00[r] = (__bf16)v00; p01[r] = (__bf16)v01;
      p10[r] = (__bf16)v10; p11[r] = (__bf16)v11;
    }
    long o00, o01, o10, o11;
    if (flags & GF_SPLITB) {
      int bb = mBase >> 9, t = mBase & 511;  // block of 64 rows never crosses 512
      o00 = ((long)bb * ND + n0) * NT + t;
      o01 = ((long)bb * ND + n1) * NT + t;
      o10 = o00 + 16; o11 = o01 + 16;
    } else {
      o00 = (long)n0 * ld2 + mBase;
      o01 = (long)n1 * ld2 + mBase;
      o10 = o00 + 16; o11 = o01 + 16;
    }
    *(v8bf*)(C2 + o00) = p00; *(v8bf*)(C2 + o01) = p01;
    *(v8bf*)(C2 + o10) = p10; *(v8bf*)(C2 + o11) = p11;
  } else {
    float* C = (float*)Cg + (long)z * sCz;
#pragma unroll
    for (int r = 0; r < 8; r++) {
      long i00 = (long)(mBase + r) * ldc + n0;
      long i01 = (long)(mBase + r) * ldc + n1;
      long i10 = (long)(mBase + 16 + r) * ldc + n0;
      long i11 = (long)(mBase + 16 + r) * ldc + n1;
      float v00 = alpha * acc00[r] + b0;
      float v01 = alpha * acc01[r] + b1;
      float v10 = alpha * acc10[r] + b0;
      float v11 = alpha * acc11[r] + b1;
      if (flags & GF_ACC) { v00 += C[i00]; v01 += C[i01]; v10 += C[i10]; v11 += C[i11]; }
      C[i00] = v00; C[i01] = v01; C[i10] = v10; C[i11] = v11;
    }
  }
}

// ---------------- scalar glue kernels ----------------
__global__ void k_seqmean_diff(const float* __restrict__ x, float* sm, float* df) {
  int i = blockIdx.x * blockDim.x + threadIdx.x;
  if (i >= NB * ND) return;
  int b = i >> 8, d = i & 255;
  float s = 0.f;
  for (int t = 0; t < NT; t++) s += x[((long)b * NT + t) * ND + d];
  sm[i] = s * (1.0f / NT);
  df[i] = (x[((long)b * NT + NT - 1) * ND + d] - x[(long)b * NT * ND + d]) * (1.0f / (NT - 1));
}

__global__ void k_gpath(const float* __restrict__ sm,
                        const float* gW1, const float* gb1,
                        const float* lnw, const float* lnb,
                        const float* gW2, const float* gb2,
                        float* gprobs) {
  __shared__ float red[256];
  __shared__ float act[NHID];
  __shared__ float gl[NGS];
  int b = blockIdx.x, j = threadIdx.x;
  float h = gb1[j];
  for (int d = 0; d < ND; d++) h += sm[b * ND + d] * gW1[d * NHID + j];
  float mean = block_sum<256>(h, red, j) * (1.0f / NHID);
  float var  = block_sum<256>((h - mean) * (h - mean), red, j) * (1.0f / NHID);
  float y = (h - mean) * rsqrtf(var + 1e-5f) * lnw[j] + lnb[j];
  act[j] = gelu_f(y);
  __syncthreads();
  if (j < NGS) {
    float s = gb2[j];
    for (int k = 0; k < NHID; k++) s += act[k] * gW2[k * NGS + j];
    gl[j] = s;
  }
  __syncthreads();
  if (j == 0) {
    float v[NGS], mx = -1e30f;
    for (int s = 0; s < NGS; s++) {
      v[s] = (gl[s] + gumbel_h(0x12340000u + (unsigned)(b * NGS + s))) * 2.0f; // /TAU
      mx = fmaxf(mx, v[s]);
    }
    float sum = 0.f;
    for (int s = 0; s < NGS; s++) { v[s] = expf(v[s] - mx); sum += v[s]; }
    for (int s = 0; s < NGS; s++) gprobs[b * NGS + s] = v[s] / sum;
  }
}

__global__ void k_linear(const float* __restrict__ X, const float* __restrict__ W,
                         const float* __restrict__ bias, float* __restrict__ Y,
                         int Kin, int Cout, int dogelu) {
  int i = blockIdx.x * blockDim.x + threadIdx.x;
  if (i >= NB * Cout) return;
  int b = i / Cout, c = i % Cout;
  float s = bias[c];
  for (int k = 0; k < Kin; k++) s += X[b * Kin + k] * W[k * Cout + c];
  Y[i] = dogelu ? gelu_f(s) : s;
}

__global__ void k_gates(const float* __restrict__ slog, const float* __restrict__ alog,
                        const float* __restrict__ gprobs, const float* __restrict__ g2f,
                        const float* __restrict__ semb, const float* __restrict__ aemb,
                        float* gate) {
  int i = blockIdx.x * blockDim.x + threadIdx.x;
  if (i >= NB * ND) return;
  int b = i >> 8, d = i & 255;
  float sv[NSS], mx = -1e30f, sum = 0.f, sg = 0.f;
  for (int s = 0; s < NSS; s++) {
    sv[s] = (slog[(b * ND + d) * NSS + s] + gumbel_h(0x51AB0000u + (unsigned)(i * 8 + s))) * 2.0f;
    mx = fmaxf(mx, sv[s]);
  }
  for (int s = 0; s < NSS; s++) { sv[s] = expf(sv[s] - mx); sum += sv[s]; }
  for (int s = 0; s < NSS; s++) sg += (sv[s] / sum) * semb[s * ND + d];
  float av[NAS]; mx = -1e30f; sum = 0.f; float ag = 0.f;
  for (int s = 0; s < NAS; s++) {
    av[s] = (alog[(b * ND + d) * NAS + s] + gumbel_h(0x77C10000u + (unsigned)(i * 8 + s))) * 2.0f;
    mx = fmaxf(mx, av[s]);
  }
  for (int s = 0; s < NAS; s++) { av[s] = expf(av[s] - mx); sum += av[s]; }
  for (int s = 0; s < NAS; s++) ag += (av[s] / sum) * aemb[s * ND + d];
  float gg = 0.f;
  for (int s = 0; s < NGS; s++) gg += gprobs[b * NGS + s] * g2f[s * ND + d];
  gate[i] = sigm_f(gg + sg + ag);
}

__global__ void k_ema(const float* __restrict__ x, const float* __restrict__ gate,
                      const float* __restrict__ df, const float* __restrict__ db,
                      float* __restrict__ hsum) {
  int i = blockIdx.x * blockDim.x + threadIdx.x;
  if (i >= NB * ND) return;
  int b = i >> 8, d = i & 255;
  float g = gate[i];
  float af = sigm_f(df[d]), ab = sigm_f(db[d]);
  float h = 0.f;
  for (int t = 0; t < NT; t++) {
    float xv = x[((long)b * NT + t) * ND + d] * g;
    h = af * h + (1.0f - af) * xv;
    hsum[((long)b * NT + t) * ND + d] = h;
  }
  h = 0.f;
  for (int t = NT - 1; t >= 0; t--) {
    float xv = x[((long)b * NT + t) * ND + d] * g;
    h = ab * h + (1.0f - ab) * xv;
    hsum[((long)b * NT + t) * ND + d] += h;
  }
}

__global__ void k_f2bf(const float* __restrict__ src, __bf16* __restrict__ dst, long n) {
  long i = (long)blockIdx.x * blockDim.x + threadIdx.x;
  if (i < n) dst[i] = (__bf16)src[i];
}

// dst[c*R + r] = (bf16) src[r*C0 + c]
__global__ void k_transpose_bf(const float* __restrict__ src, __bf16* __restrict__ dst,
                               int R, int C0) {
  int i = blockIdx.x * blockDim.x + threadIdx.x;
  if (i >= R * C0) return;
  int r = i / C0, c = i % C0;
  dst[(long)c * R + r] = (__bf16)src[(long)r * C0 + c];
}

__global__ void k_adj_logits(const float* __restrict__ psi, float* __restrict__ lg) {
  int i = blockIdx.x * blockDim.x + threadIdx.x;
  if (i >= NH * ND * ND) return;
  int h = i >> 16, n = (i >> 8) & 255, m = i & 255;
  float s = 0.f;
  for (int e = 0; e < NDE; e++)
    s += psi[(h * ND + n) * NDE + e] * psi[(h * ND + m) * NDE + e];
  lg[i] = s * 0.31622776601f; // 1/sqrt(10)
}

__global__ void k_adj_sample(const float* __restrict__ lg, const float* __restrict__ lsig,
                             float* __restrict__ A) {
  int i = blockIdx.x * blockDim.x + threadIdx.x;
  if (i >= NMC * NH * ND * ND) return;
  int rem = i & (NH * ND * ND - 1);
  int h = (i >> 16) & 3;
  A[i] = lg[rem] + expf(lsig[h]) * gauss_h(0xA511E9B3u ^ (unsigned)(i * 2654435761u));
}

__global__ void k_row_softmax(float* __restrict__ A) {
  __shared__ float red[256];
  int row = blockIdx.x, tid = threadIdx.x;
  float v = A[(long)row * ND + tid];
  float mx = block_max<256>(v, red, tid);
  float e = expf(v - mx);
  float s = block_sum<256>(e, red, tid);
  A[(long)row * ND + tid] = e / s;
}

__global__ void k_ctx(const float* __restrict__ gate, float* __restrict__ ctx) {
  __shared__ float red[256];
  int d = threadIdx.x;
  float c = 0.f;
  for (int b = 0; b < NB; b++) c += gate[b * ND + d];
  c *= (1.0f / NB);
  float n2 = block_sum<256>(c * c, red, d);
  float cn = c / (sqrtf(n2) + 1e-6f);
  ctx[d] = cn;
  float cs = block_sum<256>(cn, red, d);
  if (d == 0) ctx[ND] = cs;
}

__global__ void k_outer(const float* __restrict__ ctx, float* __restrict__ outer) {
  int i = blockIdx.x * blockDim.x + threadIdx.x;
  if (i >= ND * ND) return;
  int n = i >> 8, m = i & 255;
  float num = ctx[n] * ctx[m];
  outer[i] = num / fmaxf(ctx[n] * ctx[ND], 1e-6f);
}

__global__ void k_diffmix(float* __restrict__ A, const float* __restrict__ outer,
                          const float* __restrict__ steplog, const float* __restrict__ nscale,
                          int s) {
  int i = blockIdx.x * blockDim.x + threadIdx.x;
  if (i >= NMC * NH * ND * ND) return;
  int n = (i >> 8) & 255, m = i & 255;
  float alpha = sigm_f(steplog[s]);
  float noise = nscale[s] * gauss_h(0xBEEF0000u + (unsigned)s * 0x01000193u + (unsigned)i);
  A[i] = alpha * A[i] + (1.0f - alpha) * outer[n * ND + m] + noise;
}

__global__ void k_symm(float* __restrict__ A) {
  int i = blockIdx.x * blockDim.x + threadIdx.x;
  if (i >= NMC * NH * ND * ND) return;
  int n = (i >> 8) & 255, m = i & 255;
  if (n < m) {
    long base = (long)(i >> 16) * ND * ND;
    float a = A[base + n * ND + m], b = A[base + m * ND + n];
    float v = 0.5f * (a + b);
    A[base + n * ND + m] = v;
    A[base + m * ND + n] = v;
  }
}

__global__ void k_heads(const float* __restrict__ Z, const float* __restrict__ mw,
                        const float* __restrict__ mb, const float* __restrict__ lw,
                        const float* __restrict__ lb, float* gm, float* gl) {
  int i = blockIdx.x * blockDim.x + threadIdx.x;
  if (i >= NB * ND) return;
  float sm = 0.f, sl = 0.f;
  const float* zp = Z + (long)i * NDH;
  for (int j = 0; j < NDH; j++) { sm += zp[j] * mw[j]; sl += zp[j] * lw[j]; }
  gm[i] = sm + mb[0];
  gl[i] = sl + lb[0];
}

__global__ void k_decoder(const float* __restrict__ gm, const float* __restrict__ gl,
                          const float* dW1, const float* db1,
                          const float* lnw, const float* lnb,
                          const float* mW, const float* mb,
                          const float* lvW, const float* lvb,
                          float* __restrict__ out) {
  __shared__ float red[NDECH];
  int b = blockIdx.x, j = threadIdx.x;
  float h = db1[j];
  for (int n = 0; n < ND; n++) h += gm[b * ND + n] * dW1[n * NDECH + j];
  float a = gelu_f(h);
  float mean = block_sum<NDECH>(a, red, j) * (1.0f / NDECH);
  float var  = block_sum<NDECH>((a - mean) * (a - mean), red, j) * (1.0f / NDECH);
  float feat = (a - mean) * rsqrtf(var + 1e-5f) * lnw[j] + lnb[j];
  float mo = block_sum<NDECH>(feat * mW[j], red, j) + mb[0];
  float lo = block_sum<NDECH>(feat * lvW[j], red, j) + lvb[0];
  float se = block_sum<NDECH>(expf(gl[b * ND + j]) + expf(gl[b * ND + j + NDECH]), red, j);
  if (j == 0) {
    out[b * 2 + 0] = mo;
    out[b * 2 + 1] = logf(expf(lo) + se);
  }
}

// ---------------- host launcher ----------------
static inline int gdim(long n) { return (int)((n + 255) / 256); }

extern "C" void kernel_launch(void* const* d_in, const int* in_sizes, int n_in,
                              void* d_out, int out_size, void* d_ws, size_t ws_size,
                              hipStream_t stream) {
  (void)in_sizes; (void)n_in; (void)out_size; (void)ws_size;
  const float* x     = (const float*)d_in[0];
  const float* gW1   = (const float*)d_in[1];
  const float* gb1   = (const float*)d_in[2];
  const float* glnw  = (const float*)d_in[3];
  const float* glnb  = (const float*)d_in[4];
  const float* gW2   = (const float*)d_in[5];
  const float* gb2   = (const float*)d_in[6];
  const float* sW1   = (const float*)d_in[7];
  const float* sb1   = (const float*)d_in[8];
  const float* sW2   = (const float*)d_in[9];
  const float* sb2   = (const float*)d_in[10];
  const float* aW1   = (const float*)d_in[11];
  const float* ab1   = (const float*)d_in[12];
  const float* aW2   = (const float*)d_in[13];
  const float* ab2   = (const float*)d_in[14];
  const float* g2f   = (const float*)d_in[18];
  const float* semb  = (const float*)d_in[19];
  const float* aemb  = (const float*)d_in[20];
  const float* dcf   = (const float*)d_in[21];
  const float* dcb   = (const float*)d_in[22];
  const float* tW    = (const float*)d_in[23];
  const float* tb    = (const float*)d_in[24];
  const float* psi   = (const float*)d_in[25];
  const float* lsig  = (const float*)d_in[26];
  const float* slogit= (const float*)d_in[27];
  const float* nscale= (const float*)d_in[28];
  const float* Wk    = (const float*)d_in[29];
  const float* mw    = (const float*)d_in[30];
  const float* mb_   = (const float*)d_in[31];
  const float* lw    = (const float*)d_in[32];
  const float* lb_   = (const float*)d_in[33];
  const float* dW1   = (const float*)d_in[34];
  const float* db1   = (const float*)d_in[35];
  const float* dlnw  = (const float*)d_in[36];
  const float* dlnb  = (const float*)d_in[37];
  const float* dmW   = (const float*)d_in[38];
  const float* dmb   = (const float*)d_in[39];
  const float* dlvW  = (const float*)d_in[40];
  const float* dlvb  = (const float*)d_in[41];

  // workspace carve
  char* w = (char*)d_ws;
  size_t off = 0;
  auto carve = [&](size_t bytes) -> void* {
    off = (off + 255) & ~(size_t)255;
    void* p = w + off;
    off += bytes;
    return p;
  };
  float* f_sm    = (float*)carve(sizeof(float) * NB * ND);
  float* f_df    = (float*)carve(sizeof(float) * NB * ND);
  float* f_gate  = (float*)carve(sizeof(float) * NB * ND);
  float* f_gpro  = (float*)carve(sizeof(float) * NB * NGS);
  float* f_shid  = (float*)carve(sizeof(float) * NB * NHID);
  float* f_ahid  = (float*)carve(sizeof(float) * NB * NHID);
  float* f_slog  = (float*)carve(sizeof(float) * NB * ND * NSS);
  float* f_alog  = (float*)carve(sizeof(float) * NB * ND * NAS);
  float* f_hsum  = (float*)carve(sizeof(float) * (long)NB * NT * ND);
  float* f_lg    = (float*)carve(sizeof(float) * NH * ND * ND);
  float* f_Adj   = (float*)carve(sizeof(float) * (long)NMC * NH * ND * ND);
  float* f_ctx   = (float*)carve(sizeof(float) * 512);
  float* f_outer = (float*)carve(sizeof(float) * ND * ND);
  float* f_Z     = (float*)carve(sizeof(float) * (long)NB * ND * NDH);
  float* f_gm    = (float*)carve(sizeof(float) * NB * ND);
  float* f_gl    = (float*)carve(sizeof(float) * NB * ND);
  __bf16* h_hsum = (__bf16*)carve(sizeof(__bf16) * (long)NB * NT * ND);
  __bf16* h_tWt  = (__bf16*)carve(sizeof(__bf16) * ND * ND);           // tW^T [n][k]
  __bf16* h_Wkt  = (__bf16*)carve(sizeof(__bf16) * (NKH + 1) * NDH * NT); // Wk^T [d][t]
  __bf16* h_Adj  = (__bf16*)carve(sizeof(__bf16) * (long)NMC * NH * ND * ND);
  __bf16* h_Q0   = (__bf16*)carve(sizeof(__bf16) * (long)NB * NT * ND); // temporal [b][t][n]
  __bf16* h_P0   = (__bf16*)carve(sizeof(__bf16) * (long)NB * ND * NT); // node-major [b][n][t]
  __bf16* h_Qa   = (__bf16*)carve(sizeof(__bf16) * (long)NB * NT * ND);
  __bf16* h_Qb   = (__bf16*)carve(sizeof(__bf16) * (long)NB * NT * ND);
  __bf16* h_Pt   = (__bf16*)carve(sizeof(__bf16) * (long)NB * ND * NT);

  // ---- regime controller ----
  k_seqmean_diff<<<gdim(NB * ND), 256, 0, stream>>>(x, f_sm, f_df);
  k_gpath<<<NB, 256, 0, stream>>>(f_sm, gW1, gb1, glnw, glnb, gW2, gb2, f_gpro);
  k_linear<<<gdim(NB * NHID), 256, 0, stream>>>(f_sm, sW1, sb1, f_shid, ND, NHID, 1);
  k_linear<<<gdim(NB * NHID), 256, 0, stream>>>(f_df, aW1, ab1, f_ahid, ND, NHID, 1);
  k_linear<<<gdim((long)NB * ND * NSS), 256, 0, stream>>>(f_shid, sW2, sb2, f_slog, NHID, ND * NSS, 0);
  k_linear<<<gdim((long)NB * ND * NAS), 256, 0, stream>>>(f_ahid, aW2, ab2, f_alog, NHID, ND * NAS, 0);
  k_gates<<<gdim(NB * ND), 256, 0, stream>>>(f_slog, f_alog, f_gpro, g2f, semb, aemb, f_gate);

  // ---- bidirectional EMA + temporal projection (WMMA) ----
  k_ema<<<gdim(NB * ND), 256, 0, stream>>>(x, f_gate, dcf, dcb, f_hsum);
  k_f2bf<<<gdim((long)NB * NT * ND), 256, 0, stream>>>(f_hsum, h_hsum, (long)NB * NT * ND);
  k_transpose_bf<<<gdim(ND * ND), 256, 0, stream>>>(tW, h_tWt, ND, ND);
  for (int k = 0; k <= NKH; k++)
    k_transpose_bf<<<gdim(NT * NDH), 256, 0, stream>>>(
        Wk + (long)k * NT * NDH, h_Wkt + (long)k * NDH * NT, NT, NDH);
  // temporal = hsum @ tW + tb; C = Q0 [b][t][n], C2 = P0 [b][n][t]
  k_wmma_gemm<<<dim3(ND / 128, (NB * NT) / 64, 1), 256, 0, stream>>>(
      h_hsum, ND, 0, h_tWt, ND, 0, h_Q0, ND, 0, h_P0, NT, 0,
      tb, 1.0f, NB * NT, ND, ND, GF_DUAL | GF_SPLITB);

  // ---- Bayesian adjacency + diffusion ----
  k_adj_logits<<<gdim(NH * ND * ND), 256, 0, stream>>>(psi, f_lg);
  k_adj_sample<<<gdim((long)NMC * NH * ND * ND), 256, 0, stream>>>(f_lg, lsig, f_Adj);
  k_row_softmax<<<NMC * NH * ND, 256, 0, stream>>>(f_Adj);
  k_ctx<<<1, 256, 0, stream>>>(f_gate, f_ctx);
  k_outer<<<gdim(ND * ND), 256, 0, stream>>>(f_ctx, f_outer);
  for (int s = 0; s < NSTEPS; s++) {
    k_diffmix<<<gdim((long)NMC * NH * ND * ND), 256, 0, stream>>>(f_Adj, f_outer, slogit, nscale, s);
    k_symm<<<gdim((long)NMC * NH * ND * ND), 256, 0, stream>>>(f_Adj);
    k_row_softmax<<<NMC * NH * ND, 256, 0, stream>>>(f_Adj);
  }
  k_f2bf<<<gdim((long)NMC * NH * ND * ND), 256, 0, stream>>>(f_Adj, h_Adj, (long)NMC * NH * ND * ND);

  // ---- K-hop graph conv (all WMMA, zero LDS transposes) ----
  // hop 0: Z = P0_flat(4096x512) @ Wk[0] via Bt = Wkt[0](128x512)
  k_wmma_gemm<<<dim3(NDH / 128, (NB * ND) / 64, 1), 256, 0, stream>>>(
      h_P0, NT, 0, h_Wkt, NT, 0, f_Z, NDH, 0, nullptr, 0, 0,
      nullptr, 1.0f, NB * ND, NDH, NT, 0);
  const long sQ = (long)NT * ND;  // per-b stride of Q buffers
  const long sP = (long)ND * NT;  // per-b stride of P buffer
  for (int mh = 0; mh < NMC * NH; mh++) {
    const __bf16* Amh = h_Adj + (long)mh * ND * ND;
    const __bf16* Qin = h_Q0;
    for (int k = 1; k <= NKH; k++) {
      __bf16* Qout = (k & 1) ? h_Qa : h_Qb;
      // Q_out[b][t][n] = sum_p Q_in[b][t][p] * Adj[n][p]  (A=Q_in, Bt=Adj native)
      // dual epilogue also writes P_t[b][n][t] as packed 16B stores
      k_wmma_gemm<<<dim3(ND / 128, NT / 64, NB), 256, 0, stream>>>(
          Qin, ND, sQ, Amh, ND, 0, Qout, ND, sQ, h_Pt, NT, sP,
          nullptr, 1.0f, NT, ND, ND, GF_DUAL);
      // Z += (1/16) * P_t_flat(4096x512) @ Wk[k] via Bt = Wkt[k]
      k_wmma_gemm<<<dim3(NDH / 128, (NB * ND) / 64, 1), 256, 0, stream>>>(
          h_Pt, NT, 0, h_Wkt + (long)k * NDH * NT, NT, 0, f_Z, NDH, 0, nullptr, 0, 0,
          nullptr, 1.0f / 16.0f, NB * ND, NDH, NT, GF_ACC);
      Qin = Qout;
    }
  }

  // ---- heads + decoder ----
  k_heads<<<gdim(NB * ND), 256, 0, stream>>>(f_Z, mw, mb_, lw, lb_, f_gm, f_gl);
  k_decoder<<<NB, NDECH, 0, stream>>>(f_gm, f_gl, dW1, db1, dlnw, dlnb, dmW, dmb, dlvW, dlvb,
                                      (float*)d_out);
}
